// GNN1D_52888227283281
// MI455X (gfx1250) — matrix operand order repaired and checked by reading, not compiled, not run on hardware
//
#include <hip/hip_runtime.h>

typedef __attribute__((ext_vector_type(16))) _Float16 v16h;
typedef __attribute__((ext_vector_type(8)))  _Float16 v8h;
typedef __attribute__((ext_vector_type(8)))  float    v8f;

#define KS0   40     // column stride (halves) for the 16(->32)-channel input buffer
#define KSA   264    // column stride (halves) for 256-channel activation buffers
#define A3S   84     // column stride (floats) for the 72(->80)-row layer-2 output
#define EPSBN 1e-5f

// ---------------------------------------------------------------------------
// Pack P1(OUT x CIN) into WMMA A-fragment order (f16):
// dst[((mt*KT+kt)*32 + lane)*16 + h], K per ISA 16-bit A 16x32 layout.
// ---------------------------------------------------------------------------
template <int CIN, int OUT>
__global__ __launch_bounds__(256) void k_pack(const float* __restrict__ P1,
                                              _Float16* __restrict__ dst)
{
    constexpr int KT = (CIN + 31) / 32;
    constexpr int MT = (OUT + 15) / 16;
    int e = blockIdx.x * 256 + threadIdx.x;
    if (e >= MT * KT * 512) return;
    int h      = e & 15;
    int lane   = (e >> 4) & 31;
    int kt     = (e >> 9) % KT;
    int mt     = (e >> 9) / KT;
    int ln     = lane & 15, laneHi = lane >> 4;
    int row    = mt * 16 + ln;
    int v      = h >> 1;
    int k      = kt * 32 + ((v >> 2) << 4) + (laneHi << 3) + ((v & 3) << 1) + (h & 1);
    float val  = (row < OUT && k < CIN) ? P1[row * CIN + k] : 0.f;
    dst[e] = (_Float16)val;
}

// ---------------------------------------------------------------------------
// Per-tile epilogue applied directly to one accumulator (bias already folded
// into the WMMA C-matrix, so only relu/stats/convert/store remain).
// ---------------------------------------------------------------------------
template <bool RELU, bool STATS, bool STORE16, bool STORE32, int HKS>
__device__ __forceinline__ void tile_epilogue(const v8f& c, int col, int Mbase, int laneHi,
    const float* scr, const float* shr,
    float* rs, float* rq, float* rm,
    _Float16* Hlds, float* H32lds)
{
    v8h hv;
    float vv[8];
    #pragma unroll
    for (int r = 0; r < 8; ++r) {
        float val = c[r];
        if (RELU) val = fmaxf(val, 0.f);
        if (STATS) { rs[r] += val; rq[r] = fmaf(val, val, rq[r]); }
        if (STORE16) { rm[r] += val; hv[r] = (_Float16)(val * scr[r] + shr[r]); }
        if (STORE32) vv[r] = val;
    }
    if (STORE16)                                        // one ds_store_b128
        *(v8h*)(Hlds + col * HKS + Mbase + (laneHi << 3)) = hv;
    if (STORE32) {                                      // two ds_store_b128
        float4* p = (float4*)(H32lds + col * A3S + Mbase + (laneHi << 3));
        p[0] = make_float4(vv[0], vv[1], vv[2], vv[3]);
        p[1] = make_float4(vv[4], vv[5], vv[6], vv[7]);
    }
}

// ---------------------------------------------------------------------------
// PE layer, column-major activations in LDS. packA = prepacked A-fragments.
// ---------------------------------------------------------------------------
template <int CIN, int OUT, int XKS, int HKS, bool RELU, bool STATS, bool STORE16, bool STORE32>
__device__ void pe_layer(const _Float16* __restrict__ packA, const float* __restrict__ P2,
                         const _Float16* Xlds, const float* xmeanLds, float* bias2,
                         _Float16* Hlds, float* HmeanLds, float* H32lds,
                         const float* bnScale, const float* bnShift,
                         float* gSum, float* gSqr, int tid)
{
    for (int o = tid; o < OUT; o += 256) {              // bias2 = 0.1 * P2 @ mean_n(X)
        float acc = 0.f;
        #pragma unroll 8
        for (int c = 0; c < CIN; ++c) acc += P2[o * CIN + c] * xmeanLds[c];
        bias2[o] = 0.1f * acc;
    }
    __syncthreads();

    const int wave = tid >> 5, lane = tid & 31;
    const int laneHi = lane >> 4, ln = lane & 15;
    constexpr int KT = (CIN + 31) / 32;
    constexpr int MT = (OUT + 15) / 16;

    for (int mt = wave; mt < MT; mt += 8) {             // wave-uniform: EXEC stays full
        const int Mbase = mt * 16;

        v16h afrag[KT];                                 // A-frags: 2x b128 global each
        #pragma unroll
        for (int kt = 0; kt < KT; ++kt)
            afrag[kt] = *(const v16h*)(packA + ((size_t)(mt * KT + kt) * 32 + lane) * 16);

        float scr[8], shr[8];
        v8f cb;                                         // bias seeds the accumulator
        #pragma unroll
        for (int r = 0; r < 8; ++r) {
            int M = Mbase + r + (laneHi << 3);
            float bias = bias2[M < OUT ? M : 0];
            cb[r] = (M < OUT) ? bias : 0.f;
            scr[r] = STORE16 ? bnScale[M < OUT ? M : 0] : 1.f;
            shr[r] = STORE16 ? bnShift[M < OUT ? M : 0] : 0.f;
        }

        float rs[8] = {0,0,0,0,0,0,0,0};
        float rq[8] = {0,0,0,0,0,0,0,0};
        float rm[8] = {0,0,0,0,0,0,0,0};

        #pragma unroll
        for (int nt = 0; nt < 8; nt += 2) {             // 2 N-tiles in flight
            const int col0 = nt * 16 + ln;
            const int col1 = col0 + 16;
            v8f c0 = cb, c1 = cb;                       // C = bias (no epilogue add)
            #pragma unroll
            for (int kt = 0; kt < KT; ++kt) {
                const _Float16* p0 = Xlds + col0 * XKS + kt * 32 + (laneHi << 3);
                const _Float16* p1 = Xlds + col1 * XKS + kt * 32 + (laneHi << 3);
                v8h l0 = *(const v8h*)p0, h0 = *(const v8h*)(p0 + 16);
                v8h l1 = *(const v8h*)p1, h1 = *(const v8h*)(p1 + 16);
                v16h b0 = __builtin_shufflevector(l0, h0, 0,1,2,3,4,5,6,7,8,9,10,11,12,13,14,15);
                v16h b1 = __builtin_shufflevector(l1, h1, 0,1,2,3,4,5,6,7,8,9,10,11,12,13,14,15);
                c0 = __builtin_amdgcn_wmma_f32_16x16x32_f16(false, afrag[kt], false, b0,
                                                            (short)0, c0, false, false);
                c1 = __builtin_amdgcn_wmma_f32_16x16x32_f16(false, afrag[kt], false, b1,
                                                            (short)0, c1, false, false);
            }
            tile_epilogue<RELU, STATS, STORE16, STORE32, HKS>(c0, col0, Mbase, laneHi,
                scr, shr, rs, rq, rm, Hlds, H32lds);
            tile_epilogue<RELU, STATS, STORE16, STORE32, HKS>(c1, col1, Mbase, laneHi,
                scr, shr, rs, rq, rm, Hlds, H32lds);
        }

        if (STATS || STORE16) {
            #pragma unroll
            for (int r = 0; r < 8; ++r)
                #pragma unroll
                for (int m = 1; m < 16; m <<= 1) {      // reduce the 16 columns per half
                    if (STATS) { rs[r] += __shfl_xor(rs[r], m);
                                 rq[r] += __shfl_xor(rq[r], m); }
                    if (STORE16) rm[r] += __shfl_xor(rm[r], m);
                }
            if (ln == 0)
                #pragma unroll
                for (int r = 0; r < 8; ++r) {
                    int M = Mbase + r + (laneHi << 3);
                    if (M < OUT) {
                        if (STATS) { atomicAdd(&gSum[M], rs[r]);
                                     atomicAdd(&gSqr[M], rq[r]); }
                        if (STORE16)
                            HmeanLds[M] = (rm[r] * (1.f / 128.f)) * scr[r] + shr[r];
                    }
                }
        }
    }
    __syncthreads();
}

// Load X0 (16x128 f32) into column-major f16 LDS (K padded to 32) + row means
__device__ void load_x0(const float* __restrict__ X, _Float16* bufX, float* xmean, int tid)
{
    if (tid < 16) xmean[tid] = 0.f;
    for (int i = tid; i < 32 * 128; i += 256) {
        int r = i >> 7, n = i & 127;
        bufX[n * KS0 + r] = (r < 16) ? (_Float16)X[r * 128 + n] : (_Float16)0.f;
    }
    __syncthreads();
    {
        int r = tid >> 4, j = tid & 15;
        const float* rp = X + r * 128 + j * 8;
        float s = 0.f;
        #pragma unroll
        for (int q = 0; q < 8; ++q) s += rp[q];
        atomicAdd(&xmean[r], s * (1.f / 128.f));
    }
    __syncthreads();
}

// ws (floats): [0:256) sum0 | [256:512) sq0 | [512:768) scale0 | [768:1024) shift0 |
// [1024:1280) sum1 | [1280:1536) sq1 | [1536:1792) scale1 | [1792:2048) shift1 |
// then packed f16 A-fragments: L0 (8192 h), L1 (65536 h), L2 (20480 h)
__global__ void k_zero(float* ws)
{
    for (int i = threadIdx.x; i < 2048; i += 256) ws[i] = 0.f;
}

__global__ void k_coef(const float* __restrict__ wv, const float* __restrict__ bv,
                       const float* __restrict__ sum, const float* __restrict__ sq,
                       float* scale, float* shift)
{
    int c = threadIdx.x;
    const float invN = 1.f / (2048.f * 128.f);
    float m = sum[c] * invN;
    float var = sq[c] * invN - m * m;
    float s = wv[c] * rsqrtf(var + EPSBN);
    scale[c] = s;
    shift[c] = bv[c] - m * s;
}

__global__ __launch_bounds__(256) void k_stats0(const float* __restrict__ A,
    const _Float16* __restrict__ F0, const float* __restrict__ P20, float* ws)
{
    __shared__ __align__(16) _Float16 bufX[128 * KS0];
    __shared__ float xmean[16];
    __shared__ float bias2[256];
    const int tid = threadIdx.x;
    load_x0(A + (size_t)blockIdx.x * 2048, bufX, xmean, tid);
    pe_layer<16, 256, KS0, KSA, true, true, false, false>(F0, P20, bufX, xmean, bias2,
        nullptr, nullptr, nullptr, nullptr, nullptr, ws + 0, ws + 256, tid);
}

__global__ __launch_bounds__(256) void k_stats1(const float* __restrict__ A,
    const _Float16* __restrict__ F0, const float* __restrict__ P20,
    const _Float16* __restrict__ F1, const float* __restrict__ P21, float* ws)
{
    __shared__ __align__(16) _Float16 bufX[128 * KS0];
    __shared__ __align__(16) _Float16 bufA[128 * KSA];
    __shared__ float xmean[16];
    __shared__ float mean1[256];
    __shared__ float bias2[256];
    const int tid = threadIdx.x;
    load_x0(A + (size_t)blockIdx.x * 2048, bufX, xmean, tid);
    pe_layer<16, 256, KS0, KSA, true, false, true, false>(F0, P20, bufX, xmean, bias2,
        bufA, mean1, nullptr, ws + 512, ws + 768, nullptr, nullptr, tid);
    pe_layer<256, 256, KSA, KSA, true, true, false, false>(F1, P21, bufA, mean1, bias2,
        nullptr, nullptr, nullptr, nullptr, nullptr, ws + 1024, ws + 1280, tid);
}

__global__ __launch_bounds__(256) void k_final(const float* __restrict__ A,
    const _Float16* __restrict__ F0, const float* __restrict__ P20,
    const _Float16* __restrict__ F1, const float* __restrict__ P21,
    const _Float16* __restrict__ F2, const float* __restrict__ P22,
    const float* __restrict__ ws, float* __restrict__ out)
{
    __shared__ __align__(16) _Float16 bufX[128 * KS0];
    __shared__ __align__(16) _Float16 bufA[128 * KSA];
    __shared__ __align__(16) _Float16 bufB[128 * KSA];
    __shared__ float xmean[16];
    __shared__ float mean1[256];
    __shared__ float mean2[256];
    __shared__ float bias2[256];
    __shared__ float ybuf[64];
    __shared__ float ampsum[8];
    const int tid = threadIdx.x;
    const int b = blockIdx.x;

    load_x0(A + (size_t)b * 2048, bufX, xmean, tid);
    pe_layer<16, 256, KS0, KSA, true, false, true, false>(F0, P20, bufX, xmean, bias2,
        bufA, mean1, nullptr, ws + 512, ws + 768, nullptr, nullptr, tid);
    pe_layer<256, 256, KSA, KSA, true, false, true, false>(F1, P21, bufA, mean1, bias2,
        bufB, mean2, nullptr, ws + 1536, ws + 1792, nullptr, nullptr, tid);
    float* a3 = reinterpret_cast<float*>(bufA);         // column-major (n*A3S + M)
    pe_layer<256, 72, KSA, KSA, false, false, false, true>(F2, P22, bufB, mean2, bias2,
        nullptr, nullptr, a3, nullptr, nullptr, nullptr, nullptr, tid);

    // ---- tail ----
    if (tid < 8) ampsum[tid] = 0.f;
    if (tid < 64) {                                     // y1/y2: means of rows 8..71
        float s = 0.f;
        for (int n = 0; n < 128; ++n) s += a3[n * A3S + 8 + tid];
        ybuf[tid] = s * (1.f / 128.f);
    }
    __syncthreads();

    float* zout = out + 131072 + (size_t)b * 1024;      // z: (B,2,4,128)
    for (int i = tid; i < 512; i += 256) {
        int s = i >> 7, n = i & 127;
        float z1 = a3[n * A3S + s], z2 = a3[n * A3S + 4 + s];
        float inv = 1.f / sqrtf(z1 * z1 + z2 * z2);
        float zn1 = z1 * inv, zn2 = z2 * inv;
        a3[n * A3S + s] = zn1;
        a3[n * A3S + 4 + s] = zn2;
        zout[s * 128 + n] = zn1;
        zout[512 + s * 128 + n] = zn2;
    }
    __syncthreads();

    for (int i = tid; i < 1024; i += 256) {             // re/im, amp row sums
        int k = i >> 7, n = i & 127;
        float re = 0.f, im = 0.f;
        #pragma unroll
        for (int s = 0; s < 4; ++s) {
            float y1 = ybuf[k * 4 + s], y2 = ybuf[32 + k * 4 + s];
            float zn1 = a3[n * A3S + s], zn2 = a3[n * A3S + 4 + s];
            re += y1 * zn1 - y2 * zn2;
            im += y1 * zn2 + y2 * zn1;
        }
        atomicAdd(&ampsum[k], re * re + im * im);
    }
    __syncthreads();

    if (tid < 64) {                                     // y/temp: (B,2,8,4)
        int k = (tid >> 2) & 7;
        float temp = sqrtf(8.f) * sqrtf(ampsum[k]);
        out[(size_t)b * 64 + tid] = ybuf[tid] / temp;
    }
}

extern "C" void kernel_launch(void* const* d_in, const int* in_sizes, int n_in,
                              void* d_out, int out_size, void* d_ws, size_t ws_size,
                              hipStream_t stream)
{
    const float* A   = (const float*)d_in[0];
    const float* P10 = (const float*)d_in[1];
    const float* P20 = (const float*)d_in[2];
    const float* P11 = (const float*)d_in[3];
    const float* P21 = (const float*)d_in[4];
    const float* P12 = (const float*)d_in[5];
    const float* P22 = (const float*)d_in[6];
    const float* w0  = (const float*)d_in[7];
    const float* b0  = (const float*)d_in[8];
    const float* w1  = (const float*)d_in[9];
    const float* b1  = (const float*)d_in[10];
    float* out = (float*)d_out;
    float* ws  = (float*)d_ws;

    _Float16* F0 = (_Float16*)(ws + 2048);
    _Float16* F1 = F0 + 8192;       // 16 mtiles * 1 kt * 512
    _Float16* F2 = F1 + 65536;      // 16 mtiles * 8 kt * 512

    hipLaunchKernelGGL(k_zero, dim3(1), dim3(256), 0, stream, ws);
    hipLaunchKernelGGL((k_pack<16, 256>),  dim3(32),  dim3(256), 0, stream, P10, F0);
    hipLaunchKernelGGL((k_pack<256, 256>), dim3(256), dim3(256), 0, stream, P11, F1);
    hipLaunchKernelGGL((k_pack<256, 72>),  dim3(80),  dim3(256), 0, stream, P12, F2);

    hipLaunchKernelGGL(k_stats0, dim3(2048), dim3(256), 0, stream, A, F0, P20, ws);
    hipLaunchKernelGGL(k_coef,   dim3(1),    dim3(256), 0, stream, w0, b0, ws + 0,    ws + 256,  ws + 512,  ws + 768);
    hipLaunchKernelGGL(k_stats1, dim3(2048), dim3(256), 0, stream, A, F0, P20, F1, P21, ws);
    hipLaunchKernelGGL(k_coef,   dim3(1),    dim3(256), 0, stream, w1, b1, ws + 1024, ws + 1280, ws + 1536, ws + 1792);
    hipLaunchKernelGGL(k_final,  dim3(2048), dim3(256), 0, stream, A, F0, P20, F1, P21, F2, P22, ws, out);
}
